// SpatialGRU_30202210026004
// MI455X (gfx1250) — matrix operand order repaired and checked by reading, not compiled, not run on hardware
//
#include <hip/hip_runtime.h>

// ---------------------------------------------------------------------------
// SpatialGRU on gfx1250 (MI455X):
//   - column-pipelined persistent kernel (32 WGs, one per grid column)
//   - per-cell GEMMs via v_wmma_f32_16x16x32_bf16 (wave32 WMMA)
//   - weights pre-packed into per-lane WMMA B-fragment order (b128 loads)
//   - tile staging via GLOBAL_LOAD_ASYNC_TO_LDS_B128 (+ s_wait_asynccnt)
//   - next-cell input tile warmed with global_prefetch_b8
//   - all LDS addressing via integer offsets off one __shared__ base so every
//     access stays addrspace(3) (ds_*), never FLAT
//   - cross-WG sync: release flag store (agent scope) + relaxed spin/acquire
// ---------------------------------------------------------------------------

typedef __attribute__((ext_vector_type(16))) __bf16   bf16x16;
typedef __attribute__((ext_vector_type(8)))  float    f32x8;
typedef __attribute__((ext_vector_type(4)))  unsigned int u32x4;

union Frag { u32x4 u[2]; bf16x16 v; };

__device__ __forceinline__ unsigned short f2bf(float f) {
    union { float f; unsigned u; } v; v.f = f;
    unsigned r = v.u + 0x7FFFu + ((v.u >> 16) & 1u);   // round-to-nearest-even
    return (unsigned short)(r >> 16);
}
__device__ __forceinline__ float bf2f(unsigned short b) {
    union { unsigned u; float f; } v; v.u = ((unsigned)b) << 16;
    return v.f;
}
__device__ __forceinline__ float fast_rcp(float x) {   // v_rcp_f32
    return __builtin_amdgcn_rcpf(x);
}

// Async copy of 16 bytes global -> LDS (GV mode, no saddr). The LDS operand is
// the wave-relative LDS byte offset (low 32 bits of a generic pointer to LDS).
__device__ __forceinline__ void async_g2l_b128(void* lds_dst, const void* gsrc) {
    unsigned           ldsa = (unsigned)(size_t)lds_dst;
    unsigned long long ga   = (unsigned long long)(size_t)gsrc;
    asm volatile("global_load_async_to_lds_b128 %0, %1, off"
                 :: "v"(ldsa), "v"(ga) : "memory");
}
__device__ __forceinline__ void wait_async0() {
    asm volatile("s_wait_asynccnt 0x0" ::: "memory");
}

// B-fragment (32x16 bf16) from pre-packed global: block = 32 lanes x 8 dwords,
// each lane reads 32 contiguous bytes (two global_load_b128).
__device__ __forceinline__ bf16x16 load_b_glb(const unsigned int* pack, int blk) {
    const unsigned int* p = pack + (blk << 8) + ((threadIdx.x & 31) << 3);
    Frag f;
    f.u[0] = *(const u32x4*)(p);
    f.u[1] = *(const u32x4*)(p + 4);
    return f.v;
}

#define WMMA_BF16(A, B, C) \
    __builtin_amdgcn_wmma_f32_16x16x32_bf16(false, (A), false, (B), (short)0, (C), false, false)

// ---------------------------------------------------------------------------
// Prep kernel 1: pack W (256x448) and [W_ij;U_rec] (256x64) to bf16 fragment
// order; zero the pipeline flags.
// ---------------------------------------------------------------------------
__global__ void kPrepW(const float* __restrict__ W, const float* __restrict__ Wij,
                       const float* __restrict__ Urec,
                       unsigned int* __restrict__ Wp, unsigned int* __restrict__ Up,
                       int* __restrict__ flags) {
    int idx = blockIdx.x * 256 + threadIdx.x;           // < 57344
    {
        int r = idx & 7, lane = (idx >> 3) & 31, b = idx >> 8;
        int kc = b / 28, nt = b % 28;
        int n = nt * 16 + (lane & 15);
        int k = kc * 32 + ((lane >> 4) << 3) + 2 * (r & 3) + ((r >> 2) << 4);
        unsigned int lo = f2bf(W[k * 448 + n]);
        unsigned int hi = f2bf(W[(k + 1) * 448 + n]);
        Wp[idx] = lo | (hi << 16);
    }
    if (idx < 8192) {
        int r = idx & 7, lane = (idx >> 3) & 31, b = idx >> 8;
        int kc = b / 4, nt = b % 4;
        int n = nt * 16 + (lane & 15);
        int k = kc * 32 + ((lane >> 4) << 3) + 2 * (r & 3) + ((r >> 2) << 4);
        float lo = (k < 64) ? Wij[k * 64 + n] : Urec[(k - 64) * 64 + n];
        int k1 = k + 1;
        float hi = (k1 < 64) ? Wij[k1 * 64 + n] : Urec[(k1 - 64) * 64 + n];
        Up[idx] = (unsigned int)f2bf(lo) | ((unsigned int)f2bf(hi) << 16);
    }
    if (idx < 1024) flags[idx] = 0;
}

// Prep kernel 2: inputs (B,C,32,32) f32 -> Xbf[t][b][c] bf16 (cell-contiguous)
__global__ void kPrepX(const float* __restrict__ in, unsigned short* __restrict__ Xbf) {
    int idx = blockIdx.x * 256 + threadIdx.x;           // < 4194304
    int c = idx & 63, b = (idx >> 6) & 63, t = idx >> 12;
    int i = t >> 5, j = t & 31;
    Xbf[idx] = f2bf(in[((b * 64 + c) * 32 + i) * 32 + j]);
}

// LDS element offsets (in ushort units) within the dynamic shared block
#define HBUF(k)   ((k) * 4608)        // 4 rotating 64x72 bf16 h buffers
#define SS_OFF    18432               // s_ij   64x72 bf16
#define SA2_OFF   23040               // r*h    64x200 bf16
#define SZ_OFF    17920               // float index base: 35840 ushorts / 2
#define HS        72                  // padded h-row stride

// ---------------------------------------------------------------------------
// Main persistent kernel: 32 WGs x 256 threads (8 wave32 waves).
// ---------------------------------------------------------------------------
__global__ __launch_bounds__(256)
void kMain(const float* __restrict__ bias,
           const unsigned int* __restrict__ Wp, const unsigned int* __restrict__ Up,
           const unsigned short* __restrict__ Xbf,
           unsigned short* __restrict__ Hg, int* __restrict__ flags,
           float* __restrict__ out) {
    extern __shared__ unsigned short smem[];
    float* smemF = (float*)smem;      // f32 view for the z-gate buffers

    const int j    = blockIdx.x;
    const int tid  = threadIdx.x;
    const int lane = tid & 31;
    const int w    = tid >> 5;
    const int colL = lane & 15;

    // rotating buffer *offsets* (integers, so LDS addrspace inference survives)
    int oTop = HBUF(0), oDiag = HBUF(1), oLeft = HBUF(2), oNew = HBUF(3);

    for (int e = tid; e < 4608; e += 256) { smem[HBUF(0) + e] = 0; smem[HBUF(1) + e] = 0; }

    // A-fragment (16x32 bf16) loader from LDS at element offset `base`,
    // padded row stride `stride`: two ds_load_b128 per call.
    auto load_a = [&](int base, int stride, int mtile, int kofs) -> bf16x16 {
        int row = mtile * 16 + (lane & 15);
        int kb  = kofs + ((lane >> 4) << 3);
        int o   = base + row * stride + kb;
        Frag f;
        f.u[0] = *(const u32x4*)(&smem[o]);
        f.u[1] = *(const u32x4*)(&smem[o + 16]);
        return f.v;
    };

    for (int i = 0; i < 32; ++i) {
        const int t = i * 32 + j;

        { // stage s_ij into LDS via async DMA (2 x 16B per thread = 8KB tile)
            const unsigned short* src = Xbf + (size_t)t * 4096;
            #pragma unroll
            for (int k2 = 0; k2 < 2; ++k2) {
                int e = (tid + 256 * k2) * 8;            // 8 bf16 = 16B, row-aligned
                async_g2l_b128(&smem[SS_OFF + (e >> 6) * HS + (e & 63)], src + e);
            }
        }
        // warm next cell's input tile while this cell computes
        if (i < 31)
            __builtin_prefetch((const char*)(Xbf + (size_t)(t + 32) * 4096) + tid * 32, 0, 3);

        if (j > 0) { // wait for left neighbor, then pull its h asynchronously
            if (tid == 0) {
                while (__hip_atomic_load(&flags[t - 1], __ATOMIC_RELAXED,
                                         __HIP_MEMORY_SCOPE_AGENT) == 0)
                    __builtin_amdgcn_s_sleep(1);
                __builtin_amdgcn_fence(__ATOMIC_ACQUIRE, "agent");
            }
            __syncthreads();
            const unsigned short* src = Hg + (size_t)(t - 1) * 4096;
            #pragma unroll
            for (int k2 = 0; k2 < 2; ++k2) {
                int e = (tid + 256 * k2) * 8;
                async_g2l_b128(&smem[oLeft + (e >> 6) * HS + (e & 63)], src + e);
            }
        } else {
            for (int e = tid; e < 4096; e += 256)
                smem[oLeft + (e >> 6) * HS + (e & 63)] = 0;
        }
        wait_async0();                                   // ASYNCcnt == 0
        __syncthreads();

        // ---------------- GEMM1: G = [h_top,h_left,h_diag,s] @ W + bias -----
        const int m1 = w >> 1, hh = w & 1;
        bf16x16 A[8];
        A[0] = load_a(oTop,   HS, m1, 0);  A[1] = load_a(oTop,   HS, m1, 32);
        A[2] = load_a(oLeft,  HS, m1, 0);  A[3] = load_a(oLeft,  HS, m1, 32);
        A[4] = load_a(oDiag,  HS, m1, 0);  A[5] = load_a(oDiag,  HS, m1, 32);
        A[6] = load_a(SS_OFF, HS, m1, 0);  A[7] = load_a(SS_OFF, HS, m1, 32);
        const int rowBase = m1 * 16 + ((lane >> 4) << 3);

        if (hh == 0) {
            // r-gate tiles (cols 0..191): sigmoid, multiply matching h, stash for GEMM2
            #pragma unroll
            for (int nt = 0; nt < 12; ++nt) {
                float bv = bias[nt * 16 + colL];
                f32x8 acc = {bv, bv, bv, bv, bv, bv, bv, bv};
                #pragma unroll
                for (int kc = 0; kc < 8; ++kc)
                    acc = WMMA_BF16(A[kc], load_b_glb(Wp, kc * 28 + nt), acc);
                int c = nt * 16 + colL;
                int hoff, cc;                             // integer select, not pointers
                if (c < 64)       { hoff = oLeft; cc = c; }
                else if (c < 128) { hoff = oTop;  cc = c - 64; }
                else              { hoff = oDiag; cc = c - 128; }
                #pragma unroll
                for (int r = 0; r < 8; ++r) {
                    float rv = fast_rcp(1.f + __expf(-acc[r]));
                    int row = rowBase + r;
                    smem[SA2_OFF + row * 200 + c] =
                        f2bf(rv * bf2f(smem[hoff + row * HS + cc]));
                }
            }
        }
        // z gates: wave-local softmax across the 4 gate groups (tiles 12+g*4+kt)
        {
            int ktLo = (hh == 0) ? 0 : 1, ktHi = (hh == 0) ? 1 : 4;
            for (int kt = ktLo; kt < ktHi; ++kt) {
                f32x8 z[4];
                #pragma unroll
                for (int g = 0; g < 4; ++g) {
                    int nt = 12 + g * 4 + kt;
                    float bv = bias[nt * 16 + colL];
                    f32x8 acc = {bv, bv, bv, bv, bv, bv, bv, bv};
                    #pragma unroll
                    for (int kc = 0; kc < 8; ++kc)
                        acc = WMMA_BF16(A[kc], load_b_glb(Wp, kc * 28 + nt), acc);
                    z[g] = acc;
                }
                int col = kt * 16 + colL;
                #pragma unroll
                for (int r = 0; r < 8; ++r) {
                    float a0 = z[0][r], a1 = z[1][r], a2 = z[2][r], a3 = z[3][r];
                    float mx = fmaxf(fmaxf(a0, a1), fmaxf(a2, a3));
                    float e0 = __expf(a0 - mx), e1 = __expf(a1 - mx);
                    float e2 = __expf(a2 - mx), e3 = __expf(a3 - mx);
                    float inv = fast_rcp(e0 + e1 + e2 + e3);
                    int row = rowBase + r;
                    smemF[SZ_OFF + 0 * 4096 + row * 64 + col] = e0 * inv;   // zi
                    smemF[SZ_OFF + 1 * 4096 + row * 64 + col] = e1 * inv;   // zl
                    smemF[SZ_OFF + 2 * 4096 + row * 64 + col] = e2 * inv;   // zt
                    smemF[SZ_OFF + 3 * 4096 + row * 64 + col] = e3 * inv;   // zd
                }
            }
        }
        __syncthreads();

        // ---------------- GEMM2: h_hat = tanh([s, r*h] @ [W_ij;U_rec] + b_ij)
        const int m2 = w & 3, nb = (w >> 2) * 2;
        bf16x16 A2[8];
        A2[0] = load_a(SS_OFF, HS, m2, 0);
        A2[1] = load_a(SS_OFF, HS, m2, 32);
        #pragma unroll
        for (int kc = 2; kc < 8; ++kc)
            A2[kc] = load_a(SA2_OFF, 200, m2, (kc - 2) * 32);
        const int rowB2 = m2 * 16 + ((lane >> 4) << 3);
        #pragma unroll
        for (int nn = 0; nn < 2; ++nn) {
            int n = nb + nn;
            float bv = bias[448 + n * 16 + colL];
            f32x8 acc = {bv, bv, bv, bv, bv, bv, bv, bv};
            #pragma unroll
            for (int kc = 0; kc < 8; ++kc)
                acc = WMMA_BF16(A2[kc], load_b_glb(Up, kc * 4 + n), acc);
            int col = n * 16 + colL;
            #pragma unroll
            for (int r = 0; r < 8; ++r) {
                int row = rowB2 + r;
                float ex   = __expf(2.f * acc[r]);
                float hhat = fmaf(-2.f, fast_rcp(ex + 1.f), 1.f);  // tanh, inf-safe
                float zi = smemF[SZ_OFF + 0 * 4096 + row * 64 + col];
                float zl = smemF[SZ_OFF + 1 * 4096 + row * 64 + col];
                float zt = smemF[SZ_OFF + 2 * 4096 + row * 64 + col];
                float zd = smemF[SZ_OFF + 3 * 4096 + row * 64 + col];
                float hl = bf2f(smem[oLeft + row * HS + col]);
                float ht = bf2f(smem[oTop  + row * HS + col]);
                float hd = bf2f(smem[oDiag + row * HS + col]);
                float h  = zl * hl + zt * ht + zd * hd + zi * hhat;
                unsigned short hbv = f2bf(h);
                smem[oNew + row * HS + col] = hbv;
                Hg[(size_t)t * 4096 + row * 64 + col] = hbv;
                if (t == 1023) out[row * 64 + col] = h;            // h_last (B,u) f32
            }
        }

        asm volatile("s_wait_storecnt 0x0" ::: "memory");           // retire Hg stores
        __syncthreads();
        if (tid == 0 && j < 31)
            __hip_atomic_store(&flags[t], 1, __ATOMIC_RELEASE, __HIP_MEMORY_SCOPE_AGENT);

        // rotate: Diag <- Left, Left <- old Diag (reload target), Top <- New
        int od = oDiag; oDiag = oLeft; oLeft = od;
        int ot = oTop;  oTop  = oNew;  oNew  = ot;
    }
}

// ---------------------------------------------------------------------------
extern "C" void kernel_launch(void* const* d_in, const int* in_sizes, int n_in,
                              void* d_out, int out_size, void* d_ws, size_t ws_size,
                              hipStream_t stream) {
    const float* inputs = (const float*)d_in[0];   // (64,64,32,32)
    const float* W      = (const float*)d_in[1];   // (256,448)
    const float* Urec   = (const float*)d_in[2];   // (192,64)
    const float* bias   = (const float*)d_in[3];   // (512,)
    const float* Wij    = (const float*)d_in[4];   // (64,64)

    char* ws = (char*)d_ws;                        // ~17 MB used
    unsigned int*   Wp    = (unsigned int*)(ws);                      // 229376 B
    unsigned int*   Up    = (unsigned int*)(ws + 229376);             //  32768 B
    unsigned short* Xbf   = (unsigned short*)(ws + 262144);           //   8 MB
    unsigned short* Hg    = (unsigned short*)(ws + 262144 + 8388608); //   8 MB
    int*            flags = (int*)(ws + 262144 + 16777216);           //   4 KB

    kPrepW<<<224, 256, 0, stream>>>(W, Wij, Urec, Wp, Up, flags);
    kPrepX<<<16384, 256, 0, stream>>>(inputs, Xbf);
    kMain<<<32, 256, 137216, stream>>>(bias, Wp, Up, Xbf, Hg, flags, (float*)d_out);
}
